// Speller_7164005449947
// MI455X (gfx1250) — matrix operand order repaired and checked by reading, not compiled, not run on hardware
//
#include <hip/hip_runtime.h>

#define BS  64
#define RAL 2048
#define LAL 250
#define CS  256
#define VOC 34
#define HFS 512
#define EMB 256

typedef __attribute__((ext_vector_type(16))) __bf16        v16bf;
typedef __attribute__((ext_vector_type(8)))  float         v8f;
typedef __attribute__((ext_vector_type(4)))  unsigned int  uint4v;

union BFPack { v16bf v; uint4v u[2]; };

__device__ __forceinline__ float bflo(unsigned int x) { return __uint_as_float(x << 16); }
__device__ __forceinline__ float bfhi(unsigned int x) { return __uint_as_float(x & 0xffff0000u); }
__device__ __forceinline__ float sigmoidf_(float x)   { return 1.0f / (1.0f + __expf(-x)); }

// ---------------------------------------------------------------------------
// Prep kernels (run once per launch)
// ---------------------------------------------------------------------------
__global__ __launch_bounds__(256) void cvt_bf16_kernel(const float* __restrict__ src,
                                                       __bf16* __restrict__ dst, int n) {
    int i = blockIdx.x * 256 + threadIdx.x;
    if (i < n) dst[i] = (__bf16)src[i];
}

// W1t[k][n] (1024 x 2048), k<512 from Wih[n][k], else Whh[n][k-512]
__global__ __launch_bounds__(256) void build_wt_kernel(const float* __restrict__ Wih,
                                                       const float* __restrict__ Whh,
                                                       __bf16* __restrict__ Wt) {
    int i = blockIdx.x * 256 + threadIdx.x;
    if (i >= 1024 * 2048) return;
    int k = i >> 11;          // / 2048
    int n = i & 2047;
    float v = (k < 512) ? Wih[n * 512 + k] : Whh[n * 512 + (k - 512)];
    Wt[i] = (__bf16)v;
}

// Wt[k*rows + n] = W[n*cols + k]   (W is rows x cols)
__global__ __launch_bounds__(256) void transpose_bf16_kernel(const float* __restrict__ W,
                                                             __bf16* __restrict__ Wt,
                                                             int rows, int cols) {
    int i = blockIdx.x * 256 + threadIdx.x;
    if (i >= rows * cols) return;
    int k = i / rows;
    int n = i % rows;
    Wt[i] = (__bf16)W[n * cols + k];
}

// emball[(t*BS + b)*EMB + c] = emb_table[y[b][t]][c]
__global__ __launch_bounds__(256) void emb_kernel(const int* __restrict__ y,
                                                  const float* __restrict__ tbl,
                                                  __bf16* __restrict__ dst) {
    int i = blockIdx.x * 256 + threadIdx.x;
    if (i >= LAL * BS * EMB) return;
    int c  = i & (EMB - 1);
    int bt = i >> 8;
    int b  = bt & (BS - 1);
    int t  = bt >> 6;
    int tok = y[b * LAL + t];
    dst[i] = (__bf16)tbl[tok * EMB + c];
}

// ---------------------------------------------------------------------------
// Fused LSTM step: gates = A(64x1024) @ Wt(1024x2048) + biases -> cell update.
// KSPLIT is compile-time: k<KSPLIT reads Alo (per-step embedding buffer),
// k>=KSPLIT reads Ahi (recurrent state buffer). Two statically-bounded loops,
// no per-chunk pointer selects in the hot WMMA stream.
// Grid 64 blocks: 4 m-tiles x 16 hidden groups of 32.
// Each block: 8 waves = 2 hidden sub-tiles x 4 gates (i,f,g,o).
// ---------------------------------------------------------------------------
template <int KSPLIT>
__global__ __launch_bounds__(256) void lstm_step_kernel(
    const __bf16* __restrict__ Alo, int alo_stride,
    const __bf16* __restrict__ Ahi, int ahi_stride,
    const __bf16* __restrict__ Wt,
    const float* __restrict__ bih, const float* __restrict__ bhh,
    float* __restrict__ cstate,
    __bf16* __restrict__ hdst1, int hd1_stride, int hd1_off,
    __bf16* __restrict__ hdst2, int hd2_stride, int hd2_off,
    float* __restrict__ hf32)
{
    __shared__ float gs[8][16][16];
    const int tid   = threadIdx.x;
    const int wave  = tid >> 5;
    const int lane  = tid & 31;
    const int m0    = (blockIdx.x & 3) << 4;
    const int hbase = (blockIdx.x >> 2) << 5;
    const int ht    = wave >> 2;   // hidden sub-tile (0..1)
    const int gate  = wave & 3;    // i,f,g,o
    const int col0  = gate * HFS + hbase + (ht << 4);

    const int mrow = m0 + (lane & 15);
    const int ka   = (lane < 16) ? 0 : 8;   // ISA 16-bit A 16x32 layout

    const __bf16* aloRow = Alo + (size_t)mrow * alo_stride + ka;
    const __bf16* ahiRow = Ahi + (size_t)mrow * ahi_stride + ka;
    const __bf16* bBase  = Wt + (size_t)ka * 2048 + col0 + (lane & 15);   // col N = lane%16, row K = k + ka

    v8f acc = {0.f, 0.f, 0.f, 0.f, 0.f, 0.f, 0.f, 0.f};

    #pragma unroll 4
    for (int k = 0; k < KSPLIT; k += 32) {
        BFPack a, b;
        a.u[0] = *(const uint4v*)(aloRow + k);
        a.u[1] = *(const uint4v*)(aloRow + k + 16);
        const __bf16* brow = bBase + (size_t)k * 2048;
        for (int s = 0; s < 8; ++s) { b.v[s] = brow[(size_t)s * 2048]; b.v[s + 8] = brow[(size_t)(s + 16) * 2048]; }
        acc = __builtin_amdgcn_wmma_f32_16x16x32_bf16(false, a.v, false, b.v,
                                                      (short)0, acc, false, false);
    }
    #pragma unroll 4
    for (int k = KSPLIT; k < 1024; k += 32) {
        BFPack a, b;
        a.u[0] = *(const uint4v*)(ahiRow + (k - KSPLIT));
        a.u[1] = *(const uint4v*)(ahiRow + (k - KSPLIT) + 16);
        const __bf16* brow = bBase + (size_t)k * 2048;
        for (int s = 0; s < 8; ++s) { b.v[s] = brow[(size_t)s * 2048]; b.v[s + 8] = brow[(size_t)(s + 16) * 2048]; }
        acc = __builtin_amdgcn_wmma_f32_16x16x32_bf16(false, a.v, false, b.v,
                                                      (short)0, acc, false, false);
    }

    {   // C/D layout: VGPR v -> M = v + 8*(lane>=16), N = lane%16
        const int cc = lane & 15;
        const int rb = (lane >> 4) << 3;
        #pragma unroll
        for (int v = 0; v < 8; ++v) gs[wave][rb + v][cc] = acc[v];
    }
    __syncthreads();

    const int ml = tid >> 4;
    const int cl = tid & 15;
    const int m  = m0 + ml;
    float* crow = cstate + (size_t)m * HFS;
    #pragma unroll
    for (int h2 = 0; h2 < 2; ++h2) {
        const int hu = hbase + (h2 << 4) + cl;
        float gi = gs[h2 * 4 + 0][ml][cl] + bih[hu]           + bhh[hu];
        float gf = gs[h2 * 4 + 1][ml][cl] + bih[HFS + hu]     + bhh[HFS + hu];
        float gg = gs[h2 * 4 + 2][ml][cl] + bih[2 * HFS + hu] + bhh[2 * HFS + hu];
        float go = gs[h2 * 4 + 3][ml][cl] + bih[3 * HFS + hu] + bhh[3 * HFS + hu];
        float cn = sigmoidf_(gf) * crow[hu] + sigmoidf_(gi) * tanhf(gg);
        float hn = sigmoidf_(go) * tanhf(cn);
        crow[hu] = cn;
        hdst1[(size_t)m * hd1_stride + hd1_off + hu] = (__bf16)hn;
        if (hdst2) hdst2[(size_t)m * hd2_stride + hd2_off + hu] = (__bf16)hn;
        if (hf32)  hf32[m * HFS + hu] = hn;
    }
}

// ---------------------------------------------------------------------------
// query = h_b(64x512) @ Wqt(512x256) + bq -> bf16. 64 wave-tiles, 8 blocks.
// ---------------------------------------------------------------------------
__global__ __launch_bounds__(256) void query_kernel(
    const __bf16* __restrict__ Hb, int hstride,
    const __bf16* __restrict__ Wqt,
    const float* __restrict__ bq,
    __bf16* __restrict__ qout)
{
    const int tid = threadIdx.x, wave = tid >> 5, lane = tid & 31;
    const int gid = blockIdx.x * 8 + wave;
    const int m0  = (gid & 3) << 4;
    const int n0  = (gid >> 2) << 4;
    const int mrow = m0 + (lane & 15);
    const int ka   = (lane < 16) ? 0 : 8;

    const __bf16* aRow  = Hb + (size_t)mrow * hstride + ka;
    const __bf16* bBase = Wqt + (size_t)ka * CS + n0 + (lane & 15);

    v8f acc = {0.f, 0.f, 0.f, 0.f, 0.f, 0.f, 0.f, 0.f};
    #pragma unroll 4
    for (int k = 0; k < HFS; k += 32) {
        BFPack a, b;
        a.u[0] = *(const uint4v*)(aRow + k);
        a.u[1] = *(const uint4v*)(aRow + k + 16);
        const __bf16* brow = bBase + (size_t)k * CS;
        for (int s = 0; s < 8; ++s) { b.v[s] = brow[(size_t)s * CS]; b.v[s + 8] = brow[(size_t)(s + 16) * CS]; }
        acc = __builtin_amdgcn_wmma_f32_16x16x32_bf16(false, a.v, false, b.v,
                                                      (short)0, acc, false, false);
    }
    const int cc = lane & 15;
    const int rb = (lane >> 4) << 3;
    #pragma unroll
    for (int v = 0; v < 8; ++v) {
        float qv = acc[v] + bq[n0 + cc];
        qout[(size_t)(m0 + rb + v) * CS + n0 + cc] = (__bf16)qv;
    }
}

// ---------------------------------------------------------------------------
// Attention: energy -> softmax -> context -> vocab output. One block per batch
// element, 512 threads. key/val are bf16 (L2-resident: 134 MB < 192 MB L2).
// ---------------------------------------------------------------------------
__global__ __launch_bounds__(512) void attn_kernel(
    const __bf16* __restrict__ keyb,
    const __bf16* __restrict__ valb,
    const float*  __restrict__ mask,
    const __bf16* __restrict__ qbuf,
    const float*  __restrict__ hbf32,
    const float*  __restrict__ Wc,
    const float*  __restrict__ bc,
    __bf16* __restrict__ ctx_dst,   // row stride 768, cols 0..255
    float* __restrict__ out, int t)
{
    __shared__ float eS[RAL];
    __shared__ float qS[CS];
    __shared__ float red[16];
    __shared__ float ctxp[4][CS];
    __shared__ float ctxS[CS];

    const int tid = threadIdx.x;
    const int b   = blockIdx.x;

    if (tid < CS) qS[tid] = (float)qbuf[b * CS + tid];
    __syncthreads();

    // energy[b,r] = key[b,r,:].q  (bf16 unpack is exact via <<16)
    const unsigned int* kb = (const unsigned int*)(keyb + (size_t)b * RAL * CS);
    #pragma unroll
    for (int i = 0; i < 4; ++i) {
        int r = tid + i * 512;
        const uint4v* kr = (const uint4v*)(kb + (size_t)r * (CS / 2));
        if (i < 3) __builtin_prefetch(kb + (size_t)(r + 512) * (CS / 2), 0, 1);
        float acc = 0.f;
        for (int c = 0; c < 128; c += 4) {
            uint4v u = kr[c >> 2];
            acc += bflo(u.x) * qS[2 * c]     + bfhi(u.x) * qS[2 * c + 1];
            acc += bflo(u.y) * qS[2 * c + 2] + bfhi(u.y) * qS[2 * c + 3];
            acc += bflo(u.z) * qS[2 * c + 4] + bfhi(u.z) * qS[2 * c + 5];
            acc += bflo(u.w) * qS[2 * c + 6] + bfhi(u.w) * qS[2 * c + 7];
        }
        eS[r] = (mask[b * RAL + r] > 0.f) ? acc : -1e9f;
    }
    __syncthreads();

    // softmax: block max
    float lm = -1e30f;
    #pragma unroll
    for (int i = 0; i < 4; ++i) lm = fmaxf(lm, eS[tid + i * 512]);
    for (int off = 16; off; off >>= 1) lm = fmaxf(lm, __shfl_xor(lm, off, 32));
    if ((tid & 31) == 0) red[tid >> 5] = lm;
    __syncthreads();
    if (tid == 0) { float mx = red[0]; for (int i = 1; i < 16; ++i) mx = fmaxf(mx, red[i]); red[0] = mx; }
    __syncthreads();
    const float gmax = red[0];

    float ls = 0.f;
    #pragma unroll
    for (int i = 0; i < 4; ++i) {
        int r = tid + i * 512;
        float e = __expf(eS[r] - gmax);
        eS[r] = e;
        ls += e;
    }
    __syncthreads();
    for (int off = 16; off; off >>= 1) ls += __shfl_xor(ls, off, 32);
    if ((tid & 31) == 0) red[tid >> 5] = ls;
    __syncthreads();
    if (tid == 0) { float s = 0.f; for (int i = 0; i < 16; ++i) s += red[i]; red[0] = s; }
    __syncthreads();
    const float inv = 1.0f / red[0];

    // context[c] = sum_r attn[r]*val[b,r,c]; threads: 128 column-pairs x 4 r-quarters
    const int cp = tid & 127;
    const int rq = tid >> 7;
    const unsigned int* vb = (const unsigned int*)(valb + (size_t)b * RAL * CS)
                             + (size_t)rq * 512 * (CS / 2) + cp;
    float s0 = 0.f, s1 = 0.f;
    for (int rr = 0; rr < 512; ++rr) {
        unsigned int u = vb[(size_t)rr * (CS / 2)];
        float a = eS[rq * 512 + rr];
        s0 += a * bflo(u);
        s1 += a * bfhi(u);
    }
    ctxp[rq][2 * cp]     = s0;
    ctxp[rq][2 * cp + 1] = s1;
    __syncthreads();
    if (tid < CS) {
        float c = (ctxp[0][tid] + ctxp[1][tid] + ctxp[2][tid] + ctxp[3][tid]) * inv;
        ctxS[tid] = c;
        ctx_dst[(size_t)b * 768 + tid] = (__bf16)c;   // feeds next step's LSTM-A input
    }
    __syncthreads();

    // vocab output: out[b,t,v] = [h_b ; ctx] . Wc[v] + bc[v]
    const int wave = tid >> 5, lane = tid & 31;
    for (int v = wave; v < VOC; v += 16) {
        float acc = 0.f;
        for (int j = lane; j < HFS + CS; j += 32) {
            float f = (j < HFS) ? hbf32[b * HFS + j] : ctxS[j - HFS];
            acc += f * Wc[v * (HFS + CS) + j];
        }
        for (int off = 16; off; off >>= 1) acc += __shfl_xor(acc, off, 32);
        if (lane == 0) out[((size_t)b * LAL + t) * VOC + v] = acc + bc[v];
    }
}

// ---------------------------------------------------------------------------
// Host launcher
// ---------------------------------------------------------------------------
extern "C" void kernel_launch(void* const* d_in, const int* in_sizes, int n_in,
                              void* d_out, int out_size, void* d_ws, size_t ws_size,
                              hipStream_t stream)
{
    (void)in_sizes; (void)n_in; (void)out_size;
    const float* key    = (const float*)d_in[0];
    const float* val    = (const float*)d_in[1];
    const int*   y      = (const int*)  d_in[2];
    const float* mask   = (const float*)d_in[3];
    const float* embtab = (const float*)d_in[4];
    const float* Wih_a  = (const float*)d_in[5];
    const float* Whh_a  = (const float*)d_in[6];
    const float* bih_a  = (const float*)d_in[7];
    const float* bhh_a  = (const float*)d_in[8];
    const float* Wih_b  = (const float*)d_in[9];
    const float* Whh_b  = (const float*)d_in[10];
    const float* bih_b  = (const float*)d_in[11];
    const float* bhh_b  = (const float*)d_in[12];
    const float* Wq     = (const float*)d_in[13];
    const float* bq     = (const float*)d_in[14];
    const float* Wc     = (const float*)d_in[15];
    const float* bc     = (const float*)d_in[16];
    float* out = (float*)d_out;

    char* ws = (char*)d_ws;
    size_t off = 0;
    auto take = [&](size_t bytes) -> char* {
        char* p = ws + off;
        off += (bytes + 255) & ~(size_t)255;
        return p;
    };
    __bf16* keyb   = (__bf16*)take((size_t)BS * RAL * CS * 2);
    __bf16* valb   = (__bf16*)take((size_t)BS * RAL * CS * 2);
    __bf16* W1t    = (__bf16*)take((size_t)1024 * 2048 * 2);
    __bf16* W2t    = (__bf16*)take((size_t)1024 * 2048 * 2);
    __bf16* Wqt    = (__bf16*)take((size_t)HFS * CS * 2);
    __bf16* emball = (__bf16*)take((size_t)LAL * BS * EMB * 2);
    __bf16* X1b[2] = { (__bf16*)take((size_t)BS * 768 * 2),
                       (__bf16*)take((size_t)BS * 768 * 2) };   // [ctx | h_a]
    __bf16* X2b[2] = { (__bf16*)take((size_t)BS * 1024 * 2),
                       (__bf16*)take((size_t)BS * 1024 * 2) };  // [h_a | h_b]
    float*  c_a    = (float*)take((size_t)BS * HFS * 4);
    float*  c_b    = (float*)take((size_t)BS * HFS * 4);
    float*  hbf    = (float*)take((size_t)BS * HFS * 4);
    __bf16* qbuf   = (__bf16*)take((size_t)BS * CS * 2);
    if (off > ws_size) return;   // insufficient scratch

    // zero recurrent state (bf16/f32 zero == all-zero bytes)
    hipMemsetAsync(X1b[0], 0, (size_t)BS * 768 * 2, stream);
    hipMemsetAsync(X1b[1], 0, (size_t)BS * 768 * 2, stream);
    hipMemsetAsync(X2b[0], 0, (size_t)BS * 1024 * 2, stream);
    hipMemsetAsync(X2b[1], 0, (size_t)BS * 1024 * 2, stream);
    hipMemsetAsync(c_a, 0, (size_t)BS * HFS * 4, stream);
    hipMemsetAsync(c_b, 0, (size_t)BS * HFS * 4, stream);

    // one-time prep: KV -> bf16 (L2-resident working set), weights -> K-major bf16
    const int nkv = BS * RAL * CS;
    cvt_bf16_kernel<<<(nkv + 255) / 256, 256, 0, stream>>>(key, keyb, nkv);
    cvt_bf16_kernel<<<(nkv + 255) / 256, 256, 0, stream>>>(val, valb, nkv);
    build_wt_kernel<<<(1024 * 2048 + 255) / 256, 256, 0, stream>>>(Wih_a, Whh_a, W1t);
    build_wt_kernel<<<(1024 * 2048 + 255) / 256, 256, 0, stream>>>(Wih_b, Whh_b, W2t);
    transpose_bf16_kernel<<<(HFS * CS + 255) / 256, 256, 0, stream>>>(Wq, Wqt, CS, HFS);
    emb_kernel<<<(LAL * BS * EMB + 255) / 256, 256, 0, stream>>>(y, embtab, emball);

    for (int t = 0; t < LAL; ++t) {
        const int p = t & 1, q = p ^ 1;
        // LSTM A: A = [emb_t (k<256) | ctx,h_a from X1[p] (k>=256)]
        lstm_step_kernel<EMB><<<64, 256, 0, stream>>>(
            emball + (size_t)t * BS * EMB, EMB,
            X1b[p], 768,
            W1t, bih_a, bhh_a, c_a,
            X1b[q], 768, 256,          // h_a -> next-step X1 buffer
            X2b[p], 1024, 0,           // h_a -> this-step X2 buffer
            nullptr);
        // LSTM B: A = [h_a | h_b] from X2[p]
        lstm_step_kernel<0><<<64, 256, 0, stream>>>(
            nullptr, 0,
            X2b[p], 1024,
            W2t, bih_b, bhh_b, c_b,
            X2b[q], 1024, 512,         // h_b -> next-step X2 buffer
            nullptr, 0, 0,
            hbf);                       // f32 h_b for vocab output
        query_kernel<<<8, 256, 0, stream>>>(X2b[q] + 512, 1024, Wqt, bq, qbuf);
        attn_kernel<<<64, 512, 0, stream>>>(keyb, valb, mask, qbuf, hbf, Wc, bc,
                                            X1b[q], out, t);
    }
}
